// MMOE_T1_13907104105037
// MI455X (gfx1250) — compile-verified, matrix operand order checked by reading, not run on hardware
//
#include <hip/hip_runtime.h>
#include <stdint.h>

typedef __attribute__((ext_vector_type(16))) __bf16 v16bf;
typedef __attribute__((ext_vector_type(8)))  float  v8f;

#define E_  8
#define DIN 512
#define H_  256
#define T_  64
#define B_  16384

static __device__ __forceinline__ unsigned short f32_to_bf16(float f) {
  unsigned u = __float_as_uint(f);
  unsigned r = u + 0x7FFFu + ((u >> 16) & 1u);   // round-to-nearest-even
  return (unsigned short)(r >> 16);
}
static __device__ __forceinline__ float bf16_lo(unsigned u) {
  return __uint_as_float(u << 16);
}
static __device__ __forceinline__ float bf16_hi(unsigned u) {
  return __uint_as_float(u & 0xFFFF0000u);
}

union FragBF {
  uint4 u[2];              // 32 bytes
  v16bf v;                 // 16 bf16
  unsigned short s[16];
};

// ---------------- pre-pass: fp32 -> bf16 pack ----------------
__global__ void cvt_f32_to_bf16(const float* __restrict__ src,
                                unsigned short* __restrict__ dst, int n) {
  for (int i = blockIdx.x * blockDim.x + threadIdx.x; i < n;
       i += gridDim.x * blockDim.x)
    dst[i] = f32_to_bf16(src[i]);
}

// ---------------- fused MMoE kernel ----------------
// block = 256 threads = 8 waves (wave32); one 16-row batch tile per block.
// wave w computes expert w's 16x256 ReLU(X We^T + be) tile in registers
// (16 accumulator tiles, v_wmma_f32_16x16x32_bf16) with a 4-slot / distance-3
// software pipeline on the B fragments; experts are gate-scaled and merged
// via ds_add_f32; waves 0..3 then run the 16x64 tower GEMM (WMMA) + sigmoid.
__global__ __launch_bounds__(256)
void mmoe_fused(const unsigned short* __restrict__ Xbf,
                const unsigned short* __restrict__ Webf,
                const unsigned short* __restrict__ Wtbf,
                const float* __restrict__ be,
                const float* __restrict__ Wg,
                const float* __restrict__ bg,
                const float* __restrict__ bt,
                float* __restrict__ out) {
  __shared__ uint4 ldsX[16 * DIN / 8];     // 16 rows x 512 bf16 = 16 KB
  __shared__ float ldsComb[16 * H_];       // fp32 combined accumulator, 16 KB
  __shared__ float ldsLogit[16 * E_];
  __shared__ float ldsGate[16 * E_];

  const int tid     = threadIdx.x;
  const int lane    = tid & 31;
  const int wv      = tid >> 5;            // wave id == expert id
  const int rowbase = blockIdx.x * 16;
  const int hi      = lane >> 4;           // lane half (ISA fragment layout)
  const int nl      = lane & 15;

  // ---- stage X tile (16 KB contiguous) into LDS, zero combined ----
  {
    const uint4* src = (const uint4*)(Xbf + (size_t)rowbase * DIN);
#pragma unroll
    for (int i = 0; i < 4; ++i) ldsX[tid + i * 256] = src[tid + i * 256];
#pragma unroll
    for (int i = 0; i < 16; ++i) ldsComb[tid + i * 256] = 0.0f;
  }
  __syncthreads();

  // ---- gate logits: wave wv -> expert wv, all 16 rows; vectorized dot ----
  {
    int r = nl;
    int khalf = hi * 256;
    const float4* wg4 = (const float4*)(Wg + wv * DIN + khalf);
    const uint4*  xr4 = &ldsX[(r * DIN + khalf) >> 3];    // 8 bf16 per uint4
    float a = 0.0f;
#pragma unroll 4
    for (int k8 = 0; k8 < 32; ++k8) {                     // 32 x 8 = 256
      uint4  xu = xr4[k8];
      float4 w0 = wg4[2 * k8];
      float4 w1 = wg4[2 * k8 + 1];
      a += bf16_lo(xu.x) * w0.x + bf16_hi(xu.x) * w0.y;
      a += bf16_lo(xu.y) * w0.z + bf16_hi(xu.y) * w0.w;
      a += bf16_lo(xu.z) * w1.x + bf16_hi(xu.z) * w1.y;
      a += bf16_lo(xu.w) * w1.z + bf16_hi(xu.w) * w1.w;
    }
    a += __shfl_xor(a, 16, 32);
    if (lane < 16) ldsLogit[r * E_ + wv] = a + bg[wv];
  }
  __syncthreads();

  // ---- wave 0 softmaxes the gate (one row per lane) ----
  if (wv == 0 && lane < 16) {
    int r = lane;
    float l[E_], mx = -1e30f;
#pragma unroll
    for (int e = 0; e < E_; ++e) { l[e] = ldsLogit[r * E_ + e]; mx = fmaxf(mx, l[e]); }
    float s = 0.0f;
#pragma unroll
    for (int e = 0; e < E_; ++e) { l[e] = __expf(l[e] - mx); s += l[e]; }
    float inv = 1.0f / s;
#pragma unroll
    for (int e = 0; e < E_; ++e) ldsGate[r * E_ + e] = l[e] * inv;
  }

  // ---- expert GEMM: 16x256 per wave, K = 512, software-pipelined WMMA ----
  v8f zero = {};
  v8f acc[16];
#pragma unroll
  for (int t = 0; t < 16; ++t) acc[t] = zero;

  // B column n of the KxN tile == We[e][t*16+n][:], contiguous bf16 rows.
  const uint4* bp_base = (const uint4*)(Webf +
      ((size_t)(wv * H_ + nl) * DIN + hi * 8));

  auto loadB = [&](FragBF& b, int t, int kb) {
    const uint4* p = bp_base + (size_t)t * 1024 + (kb >> 3);
    b.u[0] = p[0];          // K {kb..kb+7}   (this lane half's chunk)
    b.u[1] = p[2];          // K {kb+16..kb+23}
  };
  auto loadA = [&](FragBF& a, int kb) {
    int aidx = (nl * DIN + kb + hi * 8) >> 3;   // uint4 units
    a.u[0] = ldsX[aidx];
    a.u[1] = ldsX[aidx + 2];
  };

  FragBF A, An, Bbuf[4];
  loadA(A, 0);
  loadB(Bbuf[0], 0, 0);
  loadB(Bbuf[1], 1, 0);
  loadB(Bbuf[2], 2, 0);

  for (int kb = 0; kb < DIN; kb += 32) {
#pragma unroll
    for (int t = 0; t < 16; ++t) {
      // prefetch the B fragment 3 tiles ahead (crossing into next kb step)
      const int tp  = (t + 3) & 15;
      const int kbp = kb + ((t >= 13) ? 32 : 0);
      if (kbp < DIN) loadB(Bbuf[(t + 3) & 3], tp, kbp);
      // prefetch next kb's A fragment mid-stream
      if (t == 8) loadA(An, (kb + 32 < DIN) ? (kb + 32) : kb);
      acc[t] = __builtin_amdgcn_wmma_f32_16x16x32_bf16(
          false, A.v, false, Bbuf[t & 3].v, (short)0, acc[t], false, false);
    }
    A = An;
  }

  __syncthreads();   // gate table complete, combined zeroed before merge

  // ---- bias + ReLU + gate scale + merge experts into LDS (ds_add_f32) ----
  {
    float g[8];
#pragma unroll
    for (int j = 0; j < 8; ++j) g[j] = ldsGate[(hi ? j + 8 : j) * E_ + wv];
#pragma unroll
    for (int t = 0; t < 16; ++t) {
      float bias = be[wv * H_ + t * 16 + nl];
#pragma unroll
      for (int j = 0; j < 8; ++j) {
        int m = hi ? j + 8 : j;                // C/D layout: VGPR j -> M=j / j+8
        float v = acc[t][j] + bias;
        v = v > 0.0f ? v : 0.0f;
        atomicAdd(&ldsComb[m * H_ + t * 16 + nl], v * g[j]);
      }
    }
  }
  __syncthreads();

  // ---- tower: waves 0..3 each produce one 16x16 output tile (K = 256) ----
  if (wv < 4) {
    const uint4* tp_base = (const uint4*)(Wtbf +
        ((size_t)(wv * 16 + nl) * H_ + hi * 8));
    auto loadBT = [&](FragBF& b, int kb) {
      const uint4* p = tp_base + (kb >> 3);
      b.u[0] = p[0];
      b.u[1] = p[2];
    };

    FragBF B2buf[2];
    loadBT(B2buf[0], 0);

    v8f acc2 = zero;
#pragma unroll
    for (int kb = 0; kb < H_; kb += 32) {
      const int slot = (kb >> 5) & 1;
      if (kb + 32 < H_) loadBT(B2buf[slot ^ 1], kb + 32);   // prefetch next
      FragBF A2;
      int koff = kb + hi * 8;
      const float* cp = &ldsComb[nl * H_ + koff];
#pragma unroll
      for (int i = 0; i < 8; ++i) {
        A2.s[i]     = f32_to_bf16(cp[i]);
        A2.s[8 + i] = f32_to_bf16(cp[16 + i]);
      }
      acc2 = __builtin_amdgcn_wmma_f32_16x16x32_bf16(
          false, A2.v, false, B2buf[slot].v, (short)0, acc2, false, false);
    }
    int col = wv * 16 + nl;
    float bias = bt[col];
#pragma unroll
    for (int j = 0; j < 8; ++j) {
      int m = hi ? j + 8 : j;
      float x = acc2[j] + bias;
      out[(size_t)(rowbase + m) * T_ + col] = 1.0f / (1.0f + __expf(-x));
    }
  }
}

extern "C" void kernel_launch(void* const* d_in, const int* in_sizes, int n_in,
                              void* d_out, int out_size, void* d_ws, size_t ws_size,
                              hipStream_t stream) {
  const float* xv = (const float*)d_in[0];
  const float* We = (const float*)d_in[1];
  const float* be = (const float*)d_in[2];
  const float* Wg = (const float*)d_in[3];
  const float* bg = (const float*)d_in[4];
  const float* Wt = (const float*)d_in[5];
  const float* bt = (const float*)d_in[6];

  // workspace layout (bf16 packs): X (16 MB) | We (2 MB) | Wt (32 KB)
  unsigned short* Xbf  = (unsigned short*)d_ws;
  unsigned short* Webf = Xbf + (size_t)B_ * DIN;
  unsigned short* Wtbf = Webf + (size_t)E_ * H_ * DIN;

  cvt_f32_to_bf16<<<2048, 256, 0, stream>>>(xv, Xbf, B_ * DIN);
  cvt_f32_to_bf16<<<512, 256, 0, stream>>>(We, Webf, E_ * H_ * DIN);
  cvt_f32_to_bf16<<<64, 256, 0, stream>>>(Wt, Wtbf, T_ * H_);

  mmoe_fused<<<B_ / 16, 256, 0, stream>>>(Xbf, Webf, Wtbf, be, Wg, bg, bt,
                                          (float*)d_out);
}
